// Simple_SHVC_57019985822312
// MI455X (gfx1250) — matrix operand order repaired and checked by reading, not compiled, not run on hardware
//
#include <hip/hip_runtime.h>
#include <hip/hip_bf16.h>
#include <math.h>

typedef __attribute__((ext_vector_type(16))) _Float16 v16h;
typedef __attribute__((ext_vector_type(8)))  float    v8f;
typedef int v4i_ __attribute__((ext_vector_type(4)));
typedef _Float16 f16;

#ifdef __has_builtin
#if __has_builtin(__builtin_amdgcn_global_load_async_to_lds_b128) && \
    __has_builtin(__builtin_amdgcn_s_wait_asynccnt)
#define USE_ASYNC_LDS 1
#endif
#endif

#define DEV static __device__ __forceinline__

DEV float softplusf(float x){ return x > 20.f ? x : log1pf(expf(x)); }
DEV float sigmoidf(float x){ return 1.f/(1.f+expf(-x)); }

DEV float block_sum256(float v){
  __shared__ float s[256];
  int t = threadIdx.x;
  s[t] = v; __syncthreads();
  for (int o = 128; o > 0; o >>= 1){ if (t < o) s[t] += s[t+o]; __syncthreads(); }
  return s[0];
}

// B-matrix fragment swizzle: logical (k, n) of B[K x NPAD] -> fragment-major
// global index so a lane's 16 halves are contiguous (two ds_load_b128).
// lane = (k%32 >= 16 ? 16 : 0) + (n%16); j = k%16; tile t = n/16; chunk kc = k/32.
DEV int swz_idx(int k, int n, int NT){
  int kc = k >> 5, kr = k & 31;
  int t  = n >> 4, nr = n & 15;
  int lane = ((kr >> 4) << 4) | nr;
  int j = kr & 15;
  return ((kc*NT + t)*32 + lane)*16 + j;
}

// ---------------------------------------------------------------------------
// WMMA GEMM: C[M x NPAD] = act(A[M x K] * B[K x NPAD] + bias)
// A f16 row-major; Bw f16 in fragment-swizzled layout (see swz_idx).
// K % 32 == 0; NPAD = NT*16; M % 256 == 0.
// Block = 8 waves; wave owns 32 rows (two 16-row A fragments) x all NPAD cols.
// B chunk (32 x NPAD, already fragment-ordered) staged in LDS per K step
// via gfx1250 async global->LDS loads when available.
// ---------------------------------------------------------------------------
template<int NT, bool DO_ELU, bool OUT_F16>
__global__ __launch_bounds__(256) void k_gemm(
    const f16* __restrict__ A, const f16* __restrict__ Bw,
    const float* __restrict__ bias, int Nreal,
    void* __restrict__ Cout, int K)
{
  constexpr int NPAD = NT*16;
  constexpr int CHUNK_V = NT*64;          // uint4 per 32xNPAD chunk
  __shared__ __align__(16) f16 sB[NT*512];
  const int lane = threadIdx.x & 31;
  const int wave = threadIdx.x >> 5;
  const int row  = lane & 15;
  const int hi   = lane >> 4;
  const int m0   = blockIdx.x*256 + wave*32;

  v8f acc[2][NT] = {};
  const f16* aRow0 = A + (size_t)(m0 + row)      * (size_t)K;
  const f16* aRow1 = A + (size_t)(m0 + 16 + row) * (size_t)K;

  const int NC = K >> 5;
  for (int kc = 0; kc < NC; ++kc) {
    const int k0 = kc << 5;
    // stage fragment-ordered B chunk into LDS (contiguous copy)
    const uint4* src = reinterpret_cast<const uint4*>(Bw + (size_t)kc * (NT*512));
    uint4* dst = reinterpret_cast<uint4*>(sB);
    for (int i = threadIdx.x; i < CHUNK_V; i += 256) {
#if defined(USE_ASYNC_LDS)
      __builtin_amdgcn_global_load_async_to_lds_b128(
          (v4i_*)(src + i), (v4i_*)(dst + i), 0, 0);
#else
      dst[i] = src[i];
#endif
    }
#if defined(USE_ASYNC_LDS)
    __builtin_amdgcn_s_wait_asynccnt(0);
#endif
    __syncthreads();

    if (kc + 1 < NC) {
      __builtin_prefetch(aRow0 + k0 + 32, 0, 1);
      __builtin_prefetch(aRow1 + k0 + 32, 0, 1);
    }

    // A fragments: lane holds row (lane&15); elems 0..15 -> k=hi*16+j
    union { uint4 u[2]; v16h h; } a0, a1;
    a0.u[0] = *reinterpret_cast<const uint4*>(aRow0 + k0 + hi*16);
    a0.u[1] = *reinterpret_cast<const uint4*>(aRow0 + k0 + hi*16 + 8);
    a1.u[0] = *reinterpret_cast<const uint4*>(aRow1 + k0 + hi*16);
    a1.u[1] = *reinterpret_cast<const uint4*>(aRow1 + k0 + hi*16 + 8);
    v16h af0 = a0.h, af1 = a1.h;

    #pragma unroll
    for (int t = 0; t < NT; ++t) {
      const f16* bp = sB + (t*32 + lane)*16;   // lane's fragment: 32 contiguous bytes
      union { uint4 u[2]; v16h h; } bu;
      bu.u[0] = *reinterpret_cast<const uint4*>(bp);
      bu.u[1] = *reinterpret_cast<const uint4*>(bp + 8);
      v16h bf = bu.h;
      acc[0][t] = __builtin_amdgcn_wmma_f32_16x16x32_f16(
          false, af0, false, bf, (short)0, acc[0][t], false, false);
      acc[1][t] = __builtin_amdgcn_wmma_f32_16x16x32_f16(
          false, af1, false, bf, (short)0, acc[1][t], false, false);
    }
    __syncthreads();
  }

  #pragma unroll
  for (int r = 0; r < 2; ++r) {
    #pragma unroll
    for (int t = 0; t < NT; ++t) {
      const int n = t*16 + row;
      const float bn = (n < Nreal) ? bias[n] : 0.f;
      #pragma unroll
      for (int v = 0; v < 8; ++v) {
        const int m = m0 + r*16 + v + 8*hi;   // C layout: VGPR v -> row v (+8 for hi lanes)
        float val = acc[r][t][v] + bn;
        if (DO_ELU) val = (val > 0.f) ? val : expm1f(val);
        if (OUT_F16) reinterpret_cast<f16*>(Cout)[(size_t)m*NPAD + n] = (f16)val;
        else         reinterpret_cast<float*>(Cout)[(size_t)m*NPAD + n] = val;
      }
    }
  }
}

static void gemm(hipStream_t st, const f16* A, const f16* Bw, const float* bias,
                 int Nreal, int Npad, void* C, int M, int K, bool elu, bool f16out)
{
  dim3 g(M/256), b(256);
  int sel = (Npad/16 - 1)*4 + (elu?2:0) + (f16out?1:0);
  switch (sel) {
    case 0:  k_gemm<1,false,false><<<g,b,0,st>>>(A,Bw,bias,Nreal,C,K); break;
    case 1:  k_gemm<1,false,true ><<<g,b,0,st>>>(A,Bw,bias,Nreal,C,K); break;
    case 2:  k_gemm<1,true ,false><<<g,b,0,st>>>(A,Bw,bias,Nreal,C,K); break;
    case 3:  k_gemm<1,true ,true ><<<g,b,0,st>>>(A,Bw,bias,Nreal,C,K); break;
    case 4:  k_gemm<2,false,false><<<g,b,0,st>>>(A,Bw,bias,Nreal,C,K); break;
    case 5:  k_gemm<2,false,true ><<<g,b,0,st>>>(A,Bw,bias,Nreal,C,K); break;
    case 6:  k_gemm<2,true ,false><<<g,b,0,st>>>(A,Bw,bias,Nreal,C,K); break;
    case 7:  k_gemm<2,true ,true ><<<g,b,0,st>>>(A,Bw,bias,Nreal,C,K); break;
    case 8:  k_gemm<3,false,false><<<g,b,0,st>>>(A,Bw,bias,Nreal,C,K); break;
    case 9:  k_gemm<3,false,true ><<<g,b,0,st>>>(A,Bw,bias,Nreal,C,K); break;
    case 10: k_gemm<3,true ,false><<<g,b,0,st>>>(A,Bw,bias,Nreal,C,K); break;
    case 11: k_gemm<3,true ,true ><<<g,b,0,st>>>(A,Bw,bias,Nreal,C,K); break;
    case 12: k_gemm<4,false,false><<<g,b,0,st>>>(A,Bw,bias,Nreal,C,K); break;
    case 13: k_gemm<4,false,true ><<<g,b,0,st>>>(A,Bw,bias,Nreal,C,K); break;
    case 14: k_gemm<4,true ,false><<<g,b,0,st>>>(A,Bw,bias,Nreal,C,K); break;
    default: k_gemm<4,true ,true ><<<g,b,0,st>>>(A,Bw,bias,Nreal,C,K); break;
  }
}

// ---------------------------------------------------------------------------
// Packing / elementwise kernels
// ---------------------------------------------------------------------------
__global__ void k_zero(float* p, int n){
  int i = blockIdx.x*blockDim.x + threadIdx.x; if (i < n) p[i] = 0.f;
}

// x[32,3,128,128] -> xd[32,12,64,64] f32 (space_to_depth)
__global__ void k_pack_xd(const float* __restrict__ x, float* __restrict__ xd, int total){
  int i = blockIdx.x*blockDim.x + threadIdx.x; if (i >= total) return;
  int b = i / 49152, r = i % 49152;
  int c = r / 4096,  p = r % 4096;
  int y = p >> 6, xx = p & 63;
  int g = c / 3, cc = c % 3;
  int dy = g >> 1, dx = g & 1;
  xd[i] = x[((b*3 + cc)*128 + 2*y + dy)*128 + 2*xx + dx];
}

// A for p_x: channels [xd[0..nxd-1], (optional up_z1 32ch), zero pad] -> [131072 x Kpad] f16
__global__ void k_pack_A_px(f16* __restrict__ A, const float* __restrict__ xd,
                            const float* __restrict__ Czup, int nxd, int Kpad, int useUp, long total){
  long i = (long)blockIdx.x*blockDim.x + threadIdx.x; if (i >= total) return;
  int k = (int)(i % Kpad); long m = i / Kpad;
  int b = (int)(m >> 12); int p = (int)(m & 4095);
  float v = 0.f;
  if (k < nxd) v = xd[((long)(b*12 + k) << 12) + p];
  else if (useUp && k < nxd + 32) {
    int c = k - nxd;
    int y = p >> 6, xx = p & 63;
    int pq = ((y & 1) << 1) | (xx & 1);
    long mu = ((long)(pq*32 + b)*32 + (y >> 1))*32 + (xx >> 1);
    v = Czup[mu*32 + c];
  }
  A[i] = (f16)v;
}

// generic NCHW f32 -> [M x Kpad] f16 (1x1 conv input)
__global__ void k_pack_A_nchw(f16* __restrict__ A, const float* __restrict__ X,
                              int Cx, int Csel, int Kpad, int HW, long total){
  long i = (long)blockIdx.x*blockDim.x + threadIdx.x; if (i >= total) return;
  int k = (int)(i % Kpad); long m = i / Kpad;
  int b = (int)(m / HW), p = (int)(m % HW);
  float v = (k < Csel) ? X[(long)(b*Cx + k)*HW + p] : 0.f;
  A[i] = (f16)v;
}

// im2col for 4x4 stride2 pad1 down-conv; src f16 position-major [B*H*W x 64]; K=1024
__global__ void k_pack_A_down(f16* __restrict__ A, const f16* __restrict__ src,
                              int H, int W, int chunkStart, long total){
  long i = (long)blockIdx.x*blockDim.x + threadIdx.x; if (i >= total) return;
  int k = (int)(i & 1023); int mo = chunkStart + (int)(i >> 10);
  int Hh = H >> 1, Wh = W >> 1;
  int b = mo / (Hh*Wh), pr = mo % (Hh*Wh);
  int h = pr / Wh, w = pr % Wh;
  int c = k >> 4, t = k & 15, ky = t >> 2, kx = t & 3;
  int iy = 2*h - 1 + ky, ix = 2*w - 1 + kx;
  f16 v = (f16)0.f;
  if (iy >= 0 && iy < H && ix >= 0 && ix < W) v = src[((long)((b*H+iy)*W+ix))*64 + c];
  A[i] = v;
}

// im2col for parity (py,px) of 4x4 transposed conv; src f16 [B*Hin*Win x 64]; K=256
__global__ void k_pack_A_up16(f16* __restrict__ A, const f16* __restrict__ src,
                              int Hin, int Win, int py, int px, long total){
  long i = (long)blockIdx.x*blockDim.x + threadIdx.x; if (i >= total) return;
  int k = (int)(i & 255); long m = i >> 8;
  int b = (int)(m / (Hin*Win)), pr = (int)(m % (Hin*Win));
  int y2 = pr / Win, x2 = pr % Win;
  int c = k >> 2, t = k & 3, ty = t >> 1, tx = t & 1;
  int iy = y2 + ty - 1 + py, ix = x2 + tx - 1 + px;
  f16 v = (f16)0.f;
  if (iy >= 0 && iy < Hin && ix >= 0 && ix < Win) v = src[((long)((b*Hin+iy)*Win+ix))*64 + c];
  A[i] = v;
}

// same, src = z1 NCHW f32 [32,32,32,32]; K=128
__global__ void k_pack_A_up32(f16* __restrict__ A, const float* __restrict__ z,
                              int py, int px, long total){
  long i = (long)blockIdx.x*blockDim.x + threadIdx.x; if (i >= total) return;
  int k = (int)(i & 127); long m = i >> 7;
  int b = (int)(m / 1024), pr = (int)(m % 1024);
  int y2 = pr >> 5, x2 = pr & 31;
  int c = k >> 2, t = k & 3, ty = t >> 1, tx = t & 1;
  int iy = y2 + ty - 1 + py, ix = x2 + tx - 1 + px;
  float v = 0.f;
  if (iy >= 0 && iy < 32 && ix >= 0 && ix < 32) v = z[((long)(b*32 + c)*32 + iy)*32 + ix];
  A[i] = (f16)v;
}

// pack w1x1 (w[o*Cin + i]) -> fragment-swizzled Bw, cols [0..kA-1] then [off..off+kB-1]
__global__ void k_pack_w1x1(f16* Bw, const float* __restrict__ w, int Cin,
                            int kA, int kB, int off, int Nreal, int Npad, int total){
  int i = blockIdx.x*blockDim.x + threadIdx.x; if (i >= total) return;
  int n = i % Npad, k = i / Npad;
  float v = 0.f;
  if (n < Nreal) {
    int col = -1;
    if (k < kA) col = k; else if (k < kA + kB) col = k - kA + off;
    if (col >= 0) v = w[n*Cin + col];
  }
  Bw[swz_idx(k, n, Npad >> 4)] = (f16)v;
}

// pack down-conv weight (cout,Cin,4,4) -> fragment-swizzled Bw, k = c*16+ky*4+kx
__global__ void k_pack_wdown(f16* Bw, const float* __restrict__ w, int Cin,
                             int Nreal, int Npad, int total){
  int i = blockIdx.x*blockDim.x + threadIdx.x; if (i >= total) return;
  int n = i % Npad, k = i / Npad;
  float v = 0.f;
  if (n < Nreal) {
    int c = k >> 4, t = k & 15, ky = t >> 2, kx = t & 3;
    v = w[((n*Cin + c)*4 + ky)*4 + kx];
  }
  Bw[swz_idx(k, n, Npad >> 4)] = (f16)v;
}

// pack up-conv weight for parity (py,px): k = c*4 + ty*2+tx, ky=2ty+py, kx=2tx+px
__global__ void k_pack_wup(f16* Bw, const float* __restrict__ w, int Cin,
                           int Nreal, int Npad, int py, int px, int total){
  int i = blockIdx.x*blockDim.x + threadIdx.x; if (i >= total) return;
  int n = i % Npad, k = i / Npad;
  float v = 0.f;
  if (n < Nreal) {
    int c = k >> 2, t = k & 3, ty = t >> 1, tx = t & 1;
    v = w[((n*Cin + c)*4 + (2*ty+py))*4 + (2*tx+px)];
  }
  Bw[swz_idx(k, n, Npad >> 4)] = (f16)v;
}

// discretized mixture-of-logistics logp; C[131072 x 16] (5 logits,5 means,5 logscales)
__global__ __launch_bounds__(256) void k_dmol(const float* __restrict__ C,
    const float* __restrict__ xd, int ych, float* logp_acc, float* save_acc){
  int m = blockIdx.x*256 + threadIdx.x;   // M = 131072 exact
  int b = m >> 12;
  float y = xd[((long)(b*12 + ych) << 12) + (m & 4095)];
  const float* pp = C + (long)m*16;
  float lg[5], lp[5];
  float mx = -1e30f;
  #pragma unroll
  for (int k = 0; k < 5; ++k){ lg[k] = pp[k]; mx = fmaxf(mx, lg[k]); }
  float se = 0.f;
  #pragma unroll
  for (int k = 0; k < 5; ++k) se += expf(lg[k]-mx);
  float lseL = mx + logf(se);
  #pragma unroll
  for (int k = 0; k < 5; ++k){
    float mean = pp[5+k];
    float lsc  = fmaxf(pp[10+k], -7.f);
    float cen  = y - mean;
    float inv  = expf(-lsc);
    float plus = inv*(cen + 1.f/255.f);
    float minn = inv*(cen - 1.f/255.f);
    float cd   = sigmoidf(plus) - sigmoidf(minn);
    float lcp  = plus - softplusf(plus);
    float lomc = -softplusf(minn);
    float mid  = inv*cen;
    float lpm  = mid - lsc - 2.f*softplusf(mid);
    float l;
    if (y < -0.999f)      l = lcp;
    else if (y > 0.999f)  l = lomc;
    else if (cd > 1e-5f)  l = logf(fmaxf(cd, 1e-12f));
    else                  l = lpm - 4.8481197f;   // log(127.5)
    lp[k] = l + (lg[k] - lseL);
  }
  float m2 = -1e30f;
  #pragma unroll
  for (int k = 0; k < 5; ++k) m2 = fmaxf(m2, lp[k]);
  float s2 = 0.f;
  #pragma unroll
  for (int k = 0; k < 5; ++k) s2 += expf(lp[k]-m2);
  float out = m2 + logf(s2);
  float bs = block_sum256(out);
  if (threadIdx.x == 0){
    atomicAdd(logp_acc + b, bs);
    if (save_acc) atomicAdd(save_acc + b, bs);
  }
}

// sample logistic + accumulate logq; C position-major [B*HW x Npad] (mu|logsd)
__global__ __launch_bounds__(256) void k_sample(const float* __restrict__ C, int Npad, int nz, int HW,
    const float* __restrict__ eps, float* __restrict__ z, float* acc){
  long i = (long)blockIdx.x*256 + threadIdx.x;   // total = B*nz*HW exact
  int b = (int)(i / ((long)nz*HW)); int r = (int)(i % ((long)nz*HW));
  int c = r / HW, p = r % HW;
  long m = (long)b*HW + p;
  float mu  = C[m*Npad + c];
  float lsd = C[m*Npad + nz + c];
  float u = eps[i];
  float zs = logf(u) - log1pf(-u);
  z[i] = mu + expf(lsd)*zs;
  float zq = -zs;
  float lp = zq - lsd - 2.f*softplusf(zq);
  float bs = block_sum256(lp);
  if (threadIdx.x == 0) atomicAdd(acc + b, bs);
}

// prior logp: C parity-ordered up-conv output, z NCHW at 2Hin x 2Win
__global__ __launch_bounds__(256) void k_prior(const float* __restrict__ C, int Npad, int nz,
    const float* __restrict__ z, int Hin, int Win, float* acc){
  long i = (long)blockIdx.x*256 + threadIdx.x;   // total = 4*32*nz*Hin*Win exact
  int per = nz*Hin*Win;
  int pb = (int)(i / per); int r = (int)(i % per);
  int pq = pb >> 5, b = pb & 31;
  int c = r / (Hin*Win); int pr = r % (Hin*Win);
  int y2 = pr / Win, x2 = pr % Win;
  int py = pq >> 1, px = pq & 1;
  long m = (long)(pq*32 + b)*(Hin*Win) + pr;
  float mu = C[m*Npad + c], lsd = C[m*Npad + nz + c];
  int H2 = 2*Hin, W2 = 2*Win;
  float zv = z[((long)(b*nz + c)*H2 + (2*y2+py))*W2 + (2*x2+px)];
  float zq = -(zv - mu)*expf(-lsd);
  float lp = zq - lsd - 2.f*softplusf(zq);
  float bs = block_sum256(lp);
  if (threadIdx.x == 0) atomicAdd(acc + b, bs);
}

// p_z3 per-step logp: C[2048 x 16] (mu=ch0, logsd=ch1), z3 NCHW [32,16,8,8]
__global__ void k_pz3(const float* __restrict__ C, const float* __restrict__ z3, int ch, float* acc){
  int m = blockIdx.x*blockDim.x + threadIdx.x;
  if (m >= 2048) return;
  int b = m >> 6, p = m & 63;
  float mu = C[(long)m*16], lsd = C[(long)m*16 + 1];
  float xv = z3[(long)(b*16 + ch)*64 + p];
  float zq = -(xv - mu)*expf(-lsd);
  float lp = zq - lsd - 2.f*softplusf(zq);
  atomicAdd(acc + b, lp);
}

__global__ void k_final(const float* __restrict__ acc, float* out){
  if (threadIdx.x == 0 && blockIdx.x == 0){
    float s1 = 0.f, s2 = 0.f;
    for (int b = 0; b < 32; ++b){
      float logp = acc[b];
      float logq = acc[32+b] + acc[96+b];      // (qz2+qz3) + init_cost
      s1 += (logq - logp);
      s2 += fmaxf(0.f, acc[64+b] - acc[96+b]); // init_save - init_cost
    }
    out[0] = (s1/32.f)*(1.4426950408889634f/49152.f) + 0.1f*(s2/32.f);
  }
}

// ---------------------------------------------------------------------------
extern "C" void kernel_launch(void* const* d_in, const int* in_sizes, int n_in,
                              void* d_out, int out_size, void* d_ws, size_t ws_size,
                              hipStream_t stream)
{
  (void)in_sizes; (void)n_in; (void)out_size; (void)ws_size;
  const float* x    = (const float*)d_in[0];
  const float* eps1 = (const float*)d_in[1];
  const float* eps2 = (const float*)d_in[2];
  const float* eps3 = (const float*)d_in[3];
  struct Net { const float *w1,*b1,*w2,*b2,*w3,*b3; };
  auto getNet = [&](int base){ Net n;
    n.w1=(const float*)d_in[base];   n.b1=(const float*)d_in[base+1];
    n.w2=(const float*)d_in[base+2]; n.b2=(const float*)d_in[base+3];
    n.w3=(const float*)d_in[base+4]; n.b3=(const float*)d_in[base+5]; return n; };
  Net p_x=getNet(4), q_z1=getNet(10), q_z2=getNet(16), q_z3=getNet(22),
      p_z1=getNet(28), p_z2=getNet(34), p_z3=getNet(40);
  const float* zup_w=(const float*)d_in[46];
  const float* zup_b=(const float*)d_in[47];

  char* wp = (char*)d_ws;
  auto carve = [&](size_t bytes)->void*{ void* p=(void*)wp; wp += (bytes+255)&~(size_t)255; return p; };
  float* acc  = (float*)carve(128*4);                  // [logp | logq | init_save | init_cost] x32
  float* xd   = (float*)carve((size_t)1572864*4);      // [32,12,64,64]
  f16*   A    = (f16*)  carve((size_t)131072*64*2);    // im2col / input buffer
  f16*   P    = (f16*)  carve((size_t)131072*64*2);    // hidden 1
  f16*   Q    = (f16*)  carve((size_t)131072*64*2);    // hidden 2
  float* C    = (float*)carve((size_t)2097152*4);      // GEMM f32 outputs
  float* Czup = (float*)carve((size_t)131072*32*4);    // up(z1), parity-ordered
  float* z1   = (float*)carve((size_t)1048576*4);
  float* z2   = (float*)carve((size_t)196608*4);
  float* z3   = (float*)carve((size_t)32768*4);
  f16*   Bw   = (f16*)  carve((size_t)1024*64*2);      // packed weights (fragment-swizzled)

  auto g1 = [](long n){ return dim3((unsigned)((n+255)/256)); };
  dim3 b256(256);

  k_zero<<<g1(128), b256, 0, stream>>>(acc, 128);
  k_pack_xd<<<g1(1572864), b256, 0, stream>>>(x, xd, 1572864);

  float *logp = acc, *logq = acc+32, *saveA = acc+64, *costA = acc+96;

  auto run_px = [&](int ych, int nxd, int Kpad, int useUp, bool save){
    int kB = useUp ? 32 : 0;
    k_pack_w1x1<<<g1(Kpad*64), b256,0,stream>>>(Bw, p_x.w1, 43, nxd, kB, 11, 64, 64, Kpad*64);
    long tA = 131072L*Kpad;
    k_pack_A_px<<<g1(tA), b256,0,stream>>>(A, xd, Czup, nxd, Kpad, useUp, tA);
    gemm(stream, A, Bw, p_x.b1, 64, 64, P, 131072, Kpad, true, true);
    k_pack_w1x1<<<g1(64*64), b256,0,stream>>>(Bw, p_x.w2, 64, 64, 0, 0, 64, 64, 64*64);
    gemm(stream, P, Bw, p_x.b2, 64, 64, Q, 131072, 64, true, true);
    k_pack_w1x1<<<g1(64*16), b256,0,stream>>>(Bw, p_x.w3, 64, 64, 0, 0, 15, 16, 64*16);
    gemm(stream, Q, Bw, p_x.b3, 15, 16, C, 131072, 64, false, false);
    k_dmol<<<dim3(512), b256, 0, stream>>>(C, xd, ych, logp, save ? saveA : nullptr);
  };

  // ---- p_x init pass (i = 1..6): predict xd[12-i] from xd[0..11-i]
  for (int i = 1; i <= 6; ++i) run_px(12-i, 12-i, 32, 0, true);

  // ---- q_z1: 1x1,1x1 at 64x64, then 4x4 stride2 down -> 64ch @ 32x32
  {
    long tA = 131072L*32;
    k_pack_A_nchw<<<g1(tA), b256,0,stream>>>(A, xd, 12, 6, 32, 4096, tA);
    k_pack_w1x1<<<g1(32*64), b256,0,stream>>>(Bw, q_z1.w1, 11, 6, 0, 0, 64, 64, 32*64);
    gemm(stream, A, Bw, q_z1.b1, 64, 64, P, 131072, 32, true, true);
    k_pack_w1x1<<<g1(64*64), b256,0,stream>>>(Bw, q_z1.w2, 64, 64, 0, 0, 64, 64, 64*64);
    gemm(stream, P, Bw, q_z1.b2, 64, 64, Q, 131072, 64, true, true);
    k_pack_wdown<<<g1(1024*64), b256,0,stream>>>(Bw, q_z1.w3, 64, 64, 64, 1024*64);
    for (int ch = 0; ch < 4; ++ch){
      int cs = ch*8192;
      long tD = 8192L*1024;
      k_pack_A_down<<<g1(tD), b256,0,stream>>>(A, Q, 64, 64, cs, tD);
      gemm(stream, A, Bw, q_z1.b3, 64, 64, C + (size_t)cs*64, 8192, 1024, false, false);
    }
    k_sample<<<g1(1048576), b256,0,stream>>>(C, 64, 32, 1024, eps1, z1, costA);
  }

  // ---- up_z1 = conv_up(z1, z_up) -> parity-ordered Czup
  for (int pq = 0; pq < 4; ++pq){
    int py = pq >> 1, px = pq & 1;
    k_pack_wup<<<g1(128*32), b256,0,stream>>>(Bw, zup_w, 32, 32, 32, py, px, 128*32);
    long tU = 32768L*128;
    k_pack_A_up32<<<g1(tU), b256,0,stream>>>(A, z1, py, px, tU);
    gemm(stream, A, Bw, zup_b, 32, 32, Czup + (size_t)pq*32768*32, 32768, 128, false, false);
  }

  // ---- p_x conditioned pass (i = 6..1): predict xd[i] from xd[0..i-1] + up_z1
  for (int i = 6; i >= 1; --i) run_px(i, i, 64, 1, false);

  // ---- q_z2: z1 -> down -> 48ch @ 16x16; sample z2
  {
    long tA = 32768L*32;
    k_pack_A_nchw<<<g1(tA), b256,0,stream>>>(A, z1, 32, 32, 32, 1024, tA);
    k_pack_w1x1<<<g1(32*64), b256,0,stream>>>(Bw, q_z2.w1, 32, 32, 0, 0, 64, 64, 32*64);
    gemm(stream, A, Bw, q_z2.b1, 64, 64, P, 32768, 32, true, true);
    k_pack_w1x1<<<g1(64*64), b256,0,stream>>>(Bw, q_z2.w2, 64, 64, 0, 0, 64, 64, 64*64);
    gemm(stream, P, Bw, q_z2.b2, 64, 64, Q, 32768, 64, true, true);
    k_pack_wdown<<<g1(1024*48), b256,0,stream>>>(Bw, q_z2.w3, 64, 48, 48, 1024*48);
    long tD = 8192L*1024;
    k_pack_A_down<<<g1(tD), b256,0,stream>>>(A, Q, 32, 32, 0, tD);
    gemm(stream, A, Bw, q_z2.b3, 48, 48, C, 8192, 1024, false, false);
    k_sample<<<g1(196608), b256,0,stream>>>(C, 48, 24, 256, eps2, z2, logq);
  }

  // ---- p_z1: z2 -> up -> 64ch @ 32x32; eval z1
  {
    long tA = 8192L*32;
    k_pack_A_nchw<<<g1(tA), b256,0,stream>>>(A, z2, 24, 24, 32, 256, tA);
    k_pack_w1x1<<<g1(32*64), b256,0,stream>>>(Bw, p_z1.w1, 24, 24, 0, 0, 64, 64, 32*64);
    gemm(stream, A, Bw, p_z1.b1, 64, 64, P, 8192, 32, true, true);
    k_pack_w1x1<<<g1(64*64), b256,0,stream>>>(Bw, p_z1.w2, 64, 64, 0, 0, 64, 64, 64*64);
    gemm(stream, P, Bw, p_z1.b2, 64, 64, Q, 8192, 64, true, true);
    for (int pq = 0; pq < 4; ++pq){
      int py = pq >> 1, px = pq & 1;
      k_pack_wup<<<g1(256*64), b256,0,stream>>>(Bw, p_z1.w3, 64, 64, 64, py, px, 256*64);
      long tU = 8192L*256;
      k_pack_A_up16<<<g1(tU), b256,0,stream>>>(A, Q, 16, 16, py, px, tU);
      gemm(stream, A, Bw, p_z1.b3, 64, 64, C + (size_t)pq*8192*64, 8192, 256, false, false);
    }
    k_prior<<<g1(1048576), b256,0,stream>>>(C, 64, 32, z1, 16, 16, logp);
  }

  // ---- q_z3: z2 -> down -> 32ch @ 8x8; sample z3
  {
    long tA = 8192L*32;
    k_pack_A_nchw<<<g1(tA), b256,0,stream>>>(A, z2, 24, 24, 32, 256, tA);
    k_pack_w1x1<<<g1(32*64), b256,0,stream>>>(Bw, q_z3.w1, 24, 24, 0, 0, 64, 64, 32*64);
    gemm(stream, A, Bw, q_z3.b1, 64, 64, P, 8192, 32, true, true);
    k_pack_w1x1<<<g1(64*64), b256,0,stream>>>(Bw, q_z3.w2, 64, 64, 0, 0, 64, 64, 64*64);
    gemm(stream, P, Bw, q_z3.b2, 64, 64, Q, 8192, 64, true, true);
    k_pack_wdown<<<g1(1024*32), b256,0,stream>>>(Bw, q_z3.w3, 64, 32, 32, 1024*32);
    long tD = 2048L*1024;
    k_pack_A_down<<<g1(tD), b256,0,stream>>>(A, Q, 16, 16, 0, tD);
    gemm(stream, A, Bw, q_z3.b3, 32, 32, C, 2048, 1024, false, false);
    k_sample<<<g1(32768), b256,0,stream>>>(C, 32, 16, 64, eps3, z3, logq);
  }

  // ---- p_z2: z3 -> up -> 48ch @ 16x16; eval z2
  {
    long tA = 2048L*32;
    k_pack_A_nchw<<<g1(tA), b256,0,stream>>>(A, z3, 16, 16, 32, 64, tA);
    k_pack_w1x1<<<g1(32*64), b256,0,stream>>>(Bw, p_z2.w1, 16, 16, 0, 0, 64, 64, 32*64);
    gemm(stream, A, Bw, p_z2.b1, 64, 64, P, 2048, 32, true, true);
    k_pack_w1x1<<<g1(64*64), b256,0,stream>>>(Bw, p_z2.w2, 64, 64, 0, 0, 64, 64, 64*64);
    gemm(stream, P, Bw, p_z2.b2, 64, 64, Q, 2048, 64, true, true);
    for (int pq = 0; pq < 4; ++pq){
      int py = pq >> 1, px = pq & 1;
      k_pack_wup<<<g1(256*48), b256,0,stream>>>(Bw, p_z2.w3, 64, 48, 48, py, px, 256*48);
      long tU = 2048L*256;
      k_pack_A_up16<<<g1(tU), b256,0,stream>>>(A, Q, 8, 8, py, px, tU);
      gemm(stream, A, Bw, p_z2.b3, 48, 48, C + (size_t)pq*2048*48, 2048, 256, false, false);
    }
    k_prior<<<g1(196608), b256,0,stream>>>(C, 48, 24, z2, 8, 8, logp);
  }

  // ---- p_z3 autoregressive channel loop (i = 1..15)
  for (int i = 1; i < 16; ++i){
    int csel = 16 - i;
    long tA = 2048L*32;
    k_pack_A_nchw<<<g1(tA), b256,0,stream>>>(A, z3, 16, csel, 32, 64, tA);
    k_pack_w1x1<<<g1(32*64), b256,0,stream>>>(Bw, p_z3.w1, 15, csel, 0, 0, 64, 64, 32*64);
    gemm(stream, A, Bw, p_z3.b1, 64, 64, P, 2048, 32, true, true);
    k_pack_w1x1<<<g1(64*64), b256,0,stream>>>(Bw, p_z3.w2, 64, 64, 0, 0, 64, 64, 64*64);
    gemm(stream, P, Bw, p_z3.b2, 64, 64, Q, 2048, 64, true, true);
    k_pack_w1x1<<<g1(64*16), b256,0,stream>>>(Bw, p_z3.w3, 64, 64, 0, 0, 2, 16, 64*16);
    gemm(stream, Q, Bw, p_z3.b3, 2, 16, C, 2048, 64, false, false);
    k_pz3<<<g1(2048), b256,0,stream>>>(C, z3, 16-i, logp);
  }

  k_final<<<dim3(1), dim3(32), 0, stream>>>(acc, (float*)d_out);
}
